// HydraNeighborhoodAttention_1700807050052
// MI455X (gfx1250) — matrix operand order repaired and checked by reading, not compiled, not run on hardware
//
#include <hip/hip_runtime.h>
#include <hip/hip_bf16.h>

typedef __attribute__((ext_vector_type(16))) __bf16 v16bf;
typedef __attribute__((ext_vector_type(8)))  float  v8f;

#define Mtot 6272   // B*H*W = 2*56*56
#define HW   3136
#define WDIM 56
#define HEADS 8
#define HD   32

// ---------------------------------------------------------------------------
// WMMA GEMM tile: C[16x16] = A[m0:m0+16, :K] * B[:K, n0:n0+16]
// A, B row-major f32; staged to LDS (b64/b128), gathered as bf16.
// Block = 256 threads (8 waves); lA shared across block, lB per-wave.
// K, N compile-time so the K-loop fully unrolls into a WMMA chain.
// ---------------------------------------------------------------------------
template <int K, int N>
__device__ __forceinline__ v8f gemm_tile_bf16(
    const float* __restrict__ A, const float* __restrict__ B,
    int m0, int n0, float* lA /*16*32*/, float* lB /*32*16, per-wave*/)
{
    const int tid  = threadIdx.x;
    const int lane = tid & 31;
    const int half = lane >> 4;
    const int l16  = lane & 15;
    v8f acc = {};
#pragma unroll
    for (int k0 = 0; k0 < K; k0 += 32) {
        // Stage A panel: 256 float2s (16 rows x 32 cols) — all threads, no divergence.
        {
            int row  = tid >> 4;            // 0..15
            int col2 = (tid & 15) << 1;     // 0,2,...,30
            ((float2*)lA)[tid] =
                *(const float2*)&A[(size_t)(m0 + row) * K + k0 + col2];
        }
        // Stage this wave's B tile: 128 float4s by 32 lanes (4 each).
#pragma unroll
        for (int r2 = 0; r2 < 4; ++r2) {
            int f4  = lane + r2 * 32;       // 0..127
            int kk  = f4 >> 2;              // 0..31
            int nn4 = (f4 & 3) << 2;        // 0,4,8,12
            ((float4*)lB)[f4] =
                *(const float4*)&B[(size_t)(k0 + kk) * N + n0 + nn4];
        }
        if (k0 + 32 < K) {
            __builtin_prefetch(&A[(size_t)m0 * K + k0 + 32], 0, 3);
            __builtin_prefetch(&B[(size_t)(k0 + 32) * N + n0], 0, 3);
        }
        __syncthreads();
        // Gather per-lane WMMA operands (ISA 7.12.2 16-bit layouts), cvt f32->bf16.
        v16bf a, b;
#pragma unroll
        for (int e = 0; e < 16; ++e) {
            int ka = (e >> 3) * 16 + half * 8 + (e & 7);   // A: M=l16, K=ka
            a[e] = (__bf16)lA[l16 * 32 + ka];
            int kb = (e >> 3) * 16 + half * 8 + (e & 7);   // B: N=l16, K=kb
            b[e] = (__bf16)lB[kb * 16 + l16];
        }
        acc = __builtin_amdgcn_wmma_f32_16x16x32_bf16(
            /*neg_a=*/false, a, /*neg_b=*/false, b,
            /*c_mod=*/(short)0, acc, /*reuse_a=*/false, /*reuse_b=*/false);
        __syncthreads();
    }
    return acc;
}

// ---------------------------------------------------------------------------
// QKV projection: qkv[m,n] = x[m,:] @ w_qkv[:,n] + b_qkv[n]; q pre-scaled.
// Scatter to ws layout [which(3)][head(8)][m(6272)][hd(32)].
// ---------------------------------------------------------------------------
__global__ __launch_bounds__(256) void qkv_gemm_kernel(
    const float* __restrict__ x, const float* __restrict__ w_qkv,
    const float* __restrict__ b_qkv, float* __restrict__ qkv)
{
    __shared__ __align__(16) float lA[16 * 32];
    __shared__ __align__(16) float lB[8][32 * 16];
    const int w  = threadIdx.x >> 5;
    const int m0 = blockIdx.x * 16;
    const int n0 = (blockIdx.y * 8 + w) * 16;
    v8f acc = gemm_tile_bf16<256, 768>(x, w_qkv, m0, n0, lA, lB[w]);

    const int lane = threadIdx.x & 31;
    const int half = lane >> 4;
    const int l16  = lane & 15;
    const float scale = 0.17677669529663687f;   // 32^-0.5
#pragma unroll
    for (int r = 0; r < 8; ++r) {
        int m = m0 + r + half * 8;
        int n = n0 + l16;
        float val = acc[r] + b_qkv[n];
        int which = n >> 8;            // 0=q, 1=k, 2=v
        if (which == 0) val *= scale;
        int head = (n >> 5) & 7;
        int d    = n & 31;
        qkv[(((size_t)which * HEADS + head) * Mtot + m) * HD + d] = val;
    }
}

// ---------------------------------------------------------------------------
// NATTEN clamped-window indices along one axis (L=56, ks=7).
// ---------------------------------------------------------------------------
__device__ __forceinline__ void nb_indices(int i, int dil, int* nb, int* bi)
{
    int r  = i % dil;
    int p  = i / dil;
    int Lr = (WDIM - r + dil - 1) / dil;
    int ps = p - 3;
    if (ps < 0) ps = 0;
    if (ps > Lr - 7) ps = Lr - 7;
#pragma unroll
    for (int a = 0; a < 7; ++a) {
        nb[a] = (ps + a) * dil + r;
        bi[a] = a + (ps - p) + 6;
    }
}

// ---------------------------------------------------------------------------
// Neighborhood attention: one thread per (b, head, i, j).
// All K/V head vectors are 32 contiguous floats -> float4 (b128) traffic.
// ---------------------------------------------------------------------------
__global__ __launch_bounds__(256) void na_attn_kernel(
    const float* __restrict__ qkv, const float* __restrict__ rpb0,
    const float* __restrict__ rpb1, float* __restrict__ attn_out)
{
    const int t    = blockIdx.x * 256 + threadIdx.x;   // exact: 196*256 = 50176
    const int j    = t % WDIM;
    const int i    = (t / WDIM) % WDIM;
    const int head = (t / HW) % HEADS;
    const int b    = t / (HW * HEADS);
    const int m    = b * HW + i * WDIM + j;

    const int split = head >> 2;                 // 0: dil=1, 1: dil=2
    const int dil   = 1 + split;
    const int hloc  = head & 3;
    const float* rpb = (split == 0 ? rpb0 : rpb1) + hloc * 13 * 13;

    int nbh[7], bih[7], nbw[7], biw[7];
    nb_indices(i, dil, nbh, bih);
    nb_indices(j, dil, nbw, biw);

    const float4* qp4 = (const float4*)(qkv + ((size_t)head * Mtot + m) * HD);
    float4 q4[8];
#pragma unroll
    for (int d4 = 0; d4 < 8; ++d4) q4[d4] = qp4[d4];

    const float* kbase = qkv + (size_t)(HEADS + head) * Mtot * HD;
    const float* vbase = qkv + (size_t)(2 * HEADS + head) * Mtot * HD;

    float lg[49];
    float mx = -3.4e38f;
#pragma unroll
    for (int a = 0; a < 7; ++a) {
#pragma unroll
        for (int c = 0; c < 7; ++c) {
            int mn = b * HW + nbh[a] * WDIM + nbw[c];
            const float4* kp4 = (const float4*)(kbase + (size_t)mn * HD);
            float s = 0.f;
#pragma unroll
            for (int d4 = 0; d4 < 8; ++d4) {
                float4 kv = kp4[d4];
                s += q4[d4].x * kv.x + q4[d4].y * kv.y +
                     q4[d4].z * kv.z + q4[d4].w * kv.w;
            }
            s += rpb[bih[a] * 13 + biw[c]];
            lg[a * 7 + c] = s;
            mx = fmaxf(mx, s);
        }
    }

    float4 acc4[8];
#pragma unroll
    for (int d4 = 0; d4 < 8; ++d4) acc4[d4] = make_float4(0.f, 0.f, 0.f, 0.f);
    float sum = 0.f;
#pragma unroll
    for (int a = 0; a < 7; ++a) {
#pragma unroll
        for (int c = 0; c < 7; ++c) {
            float wgt = __expf(lg[a * 7 + c] - mx);
            sum += wgt;
            int mn = b * HW + nbh[a] * WDIM + nbw[c];
            const float4* vp4 = (const float4*)(vbase + (size_t)mn * HD);
#pragma unroll
            for (int d4 = 0; d4 < 8; ++d4) {
                float4 vv = vp4[d4];
                acc4[d4].x += wgt * vv.x;
                acc4[d4].y += wgt * vv.y;
                acc4[d4].z += wgt * vv.z;
                acc4[d4].w += wgt * vv.w;
            }
        }
    }
    const float inv = 1.0f / sum;
    float4* op4 = (float4*)(attn_out + (size_t)m * 256 + head * HD);
#pragma unroll
    for (int d4 = 0; d4 < 8; ++d4) {
        float4 o = acc4[d4];
        op4[d4] = make_float4(o.x * inv, o.y * inv, o.z * inv, o.w * inv);
    }
}

// ---------------------------------------------------------------------------
// Output projection: out[m,n] = attn[m,:] @ w_proj[:,n] + b_proj[n]
// ---------------------------------------------------------------------------
__global__ __launch_bounds__(256) void proj_gemm_kernel(
    const float* __restrict__ attn, const float* __restrict__ w_proj,
    const float* __restrict__ b_proj, float* __restrict__ out)
{
    __shared__ __align__(16) float lA[16 * 32];
    __shared__ __align__(16) float lB[8][32 * 16];
    const int w  = threadIdx.x >> 5;
    const int m0 = blockIdx.x * 16;
    const int n0 = (blockIdx.y * 8 + w) * 16;
    v8f acc = gemm_tile_bf16<256, 256>(attn, w_proj, m0, n0, lA, lB[w]);

    const int lane = threadIdx.x & 31;
    const int half = lane >> 4;
    const int l16  = lane & 15;
#pragma unroll
    for (int r = 0; r < 8; ++r) {
        int m = m0 + r + half * 8;
        int n = n0 + l16;
        out[(size_t)m * 256 + n] = acc[r] + b_proj[n];
    }
}

extern "C" void kernel_launch(void* const* d_in, const int* in_sizes, int n_in,
                              void* d_out, int out_size, void* d_ws, size_t ws_size,
                              hipStream_t stream)
{
    const float* x      = (const float*)d_in[0];
    const float* w_qkv  = (const float*)d_in[1];
    const float* b_qkv  = (const float*)d_in[2];
    const float* w_proj = (const float*)d_in[3];
    const float* b_proj = (const float*)d_in[4];
    const float* rpb0   = (const float*)d_in[5];
    const float* rpb1   = (const float*)d_in[6];
    float* out = (float*)d_out;

    float* qkv  = (float*)d_ws;                          // 3*8*6272*32 f32 = 19.3 MB
    float* attn = qkv + (size_t)3 * HEADS * Mtot * HD;   // 6272*256 f32 = 6.4 MB

    qkv_gemm_kernel<<<dim3(Mtot / 16, 768 / 128), 256, 0, stream>>>(x, w_qkv, b_qkv, qkv);
    na_attn_kernel<<<dim3((2 * HEADS * HW) / 256), 256, 0, stream>>>(qkv, rpb0, rpb1, attn);
    proj_gemm_kernel<<<dim3(Mtot / 16, 256 / 128), 256, 0, stream>>>(attn, w_proj, b_proj, out);
}
